// GAT_Classifier_34815004901850
// MI455X (gfx1250) — matrix operand order repaired and checked by reading, not compiled, not run on hardware
//
#include <hip/hip_runtime.h>
#include <hip/hip_fp16.h>
#include <math.h>

#define N_NODES_C 100000
#define N_EDGES_C 1600000
#define N_GRAPHS_C 512
#define NODE_F_C 4
#define GLOB_F_C 10
#define HID_C 32
#define HEADS_C 4
#define NEG_SLOPE_C 0.2f
#define BN_EPS_C 1e-5f

typedef __attribute__((ext_vector_type(16))) _Float16 v16h;
typedef __attribute__((ext_vector_type(8)))  _Float16 v8h;
typedef __attribute__((ext_vector_type(8)))  float    v8f;
typedef int v4i_g __attribute__((vector_size(16)));   // matches builtin param type

#ifndef __has_builtin
#define __has_builtin(x) 0
#endif

#if defined(__AMDGCN__) && __has_builtin(__builtin_amdgcn_global_load_async_to_lds_b128)
#define HAVE_ASYNC_LDS 1
#define AS1 __attribute__((address_space(1)))
#define AS3 __attribute__((address_space(3)))
#else
#define HAVE_ASYNC_LDS 0
#endif

// ---- ordered-int mapping for float atomic max (monotonic) ----
__device__ __forceinline__ int f2ord(float f) {
  int i = __float_as_int(f);
  return i >= 0 ? i : (i ^ 0x7FFFFFFF);
}
__device__ __forceinline__ float ord2f(int i) {
  return __int_as_float(i >= 0 ? i : (i ^ 0x7FFFFFFF));
}
#define ORD_NEG_INF ((int)0x807FFFFF)

// =================== fills ===================
__global__ void k_fill_f32(float* p, long long n, float v) {
  long long i = (long long)blockIdx.x * blockDim.x + threadIdx.x;
  if (i < n) p[i] = v;
}
__global__ void k_fill_i32(int* p, long long n, int v) {
  long long i = (long long)blockIdx.x * blockDim.x + threadIdx.x;
  if (i < n) p[i] = v;
}

// =================== conversions ===================
// A: [n, inK] f32 -> [n, padK] f16 (zero-padded K)
__global__ void k_cvt_a(const float* __restrict__ src, _Float16* __restrict__ dst,
                        int n, int inK, int padK) {
  long long i = (long long)blockIdx.x * blockDim.x + threadIdx.x;
  long long tot = (long long)n * padK;
  if (i >= tot) return;
  int k = (int)(i % padK);
  long long r = i / padK;
  dst[i] = (k < inK) ? (_Float16)src[r * inK + k] : (_Float16)0.0f;
}
// W: [K, Out] f32 -> Wt: [Out, padK] f16 (transposed, zero-padded K)
__global__ void k_cvt_w(const float* __restrict__ W, _Float16* __restrict__ Wt,
                        int K, int Out, int padK) {
  int i = blockIdx.x * blockDim.x + threadIdx.x;
  int tot = Out * padK;
  if (i >= tot) return;
  int k = i % padK, o = i / padK;
  Wt[i] = (k < K) ? (_Float16)W[(long long)k * Out + o] : (_Float16)0.0f;
}

// =================== WMMA GEMM with LDS-staged B ===================
// C[N,OUT] = A16[N,K] @ B  (B supplied transposed as Bt[OUT,K], f16)
// CONTRACT: N must be a multiple of 16 (true here: 100000 = 6250*16), so
// every M-tile is full and the epilogue needs no bounds checks.
// Grid: (ceil(tilesM/8), OUT/(16*NT)). Each block stages its (16*NT) x K
// B-slice into LDS (async global->LDS when available), then each of the 8
// waves computes one 16 x (16*NT) tile. K and OUT are compile-time constants
// so the K-loop fully unrolls and all store addressing constant-folds.
template <int NT, int K, int OUT>
__global__ __launch_bounds__(256) void k_gemm_wmma(
    const _Float16* __restrict__ A, const _Float16* __restrict__ Bt,
    float* __restrict__ C, int N) {
  constexpr int COLS = 16 * NT;
  constexpr int KP = K + 8;  // pad LDS row stride to spread banks
  __shared__ _Float16 shB[COLS * KP];
  const int lane = threadIdx.x & 31;
  const int wid  = threadIdx.x >> 5;
  const int tn   = blockIdx.y;

  // ---- cooperative stage of B slice (COLS x K halves) into LDS ----
  {
    constexpr int CH = COLS * K / 8;  // 16-byte chunks
    const _Float16* src0 = Bt + (long long)tn * COLS * K;
    for (int c = threadIdx.x; c < CH; c += 256) {
      const int col = c / (K / 8);
      const int k   = (c % (K / 8)) * 8;
#if HAVE_ASYNC_LDS
      __builtin_amdgcn_global_load_async_to_lds_b128(
          (AS1 v4i_g*)(const void*)(src0 + (long long)col * K + k),
          (AS3 v4i_g*)(void*)&shB[col * KP + k], 0, 0);
#else
      *(v8h*)&shB[col * KP + k] = *(const v8h*)(src0 + (long long)col * K + k);
#endif
    }
#if HAVE_ASYNC_LDS
#if __has_builtin(__builtin_amdgcn_s_wait_asynccnt)
    __builtin_amdgcn_s_wait_asynccnt(0);
#else
    asm volatile("s_wait_asynccnt 0x0" ::: "memory");
#endif
#endif
  }
  __syncthreads();

  const int tilesM = N >> 4;  // N % 16 == 0 by contract
  const int tm = blockIdx.x * 8 + wid;
  if (tm >= tilesM) return;  // wave-uniform: EXEC stays full for WMMA

  const int nl = lane & 15;
  const int hi = lane >> 4;
  const int m = tm * 16 + nl;  // always in range (full tiles only)
  const _Float16* Arow = A + (long long)m * K + hi * 8;

  v8f acc[NT];
#pragma unroll
  for (int j = 0; j < NT; j++) { v8f z = {}; acc[j] = z; }

#pragma unroll
  for (int k0 = 0; k0 < K; k0 += 32) {
    // A fragment (ISA 16-bit A 16x32 layout): lane nl holds M=nl,
    // halves [0..7] -> K = k0 + hi*8 + 0..7 ; [8..15] -> K = k0+16 + hi*8 + 0..7
    v8h a0 = *(const v8h*)(Arow + k0);
    v8h a1 = *(const v8h*)(Arow + k0 + 16);
    v16h a;
#pragma unroll
    for (int i = 0; i < 8; i++) { a[i] = a0[i]; a[8 + i] = a1[i]; }
#pragma unroll
    for (int j = 0; j < NT; j++) {
      // B fragment from LDS: halves [0..7] -> K = k0 + hi*8 + 0..7 ;
      // [8..15] -> K = k0+16 + hi*8 + 0..7, column = j*16 + nl
      const _Float16* Bp = &shB[(j * 16 + nl) * KP + hi * 8 + k0];
      v8h b0 = *(const v8h*)(Bp);
      v8h b1 = *(const v8h*)(Bp + 16);
      v16h b;
#pragma unroll
      for (int i = 0; i < 8; i++) { b[i] = b0[i]; b[8 + i] = b1[i]; }
      acc[j] = __builtin_amdgcn_wmma_f32_16x16x32_f16(
          false, a, false, b, (short)0, acc[j], false, false);
    }
  }

  // ---- store: C/D layout VGPR r -> M = tm*16 + r + hi*8, N = nl ----
  // Full tiles only: straight-line stores, constant row strides fold into
  // immediate offsets off one base address per j.
  const int mbase = tm * 16 + hi * 8;
#pragma unroll
  for (int j = 0; j < NT; j++) {
    const int n = tn * COLS + j * 16 + nl;
    float* p = C + (long long)mbase * OUT + n;
#pragma unroll
    for (int r = 0; r < 8; r++) p[r * OUT] = acc[j][r];
  }
}

// =================== attention scores ===================
// out[n*H+h] = sum_c xs[n,h,c] * att[h,c]
__global__ void k_att(const float* __restrict__ xs, const float* __restrict__ att,
                      float* __restrict__ out, int n, int H, int C) {
  int i = blockIdx.x * blockDim.x + threadIdx.x;
  if (i >= n * H) return;
  int h = i % H;
  long long node = i / H;
  const float* xr = xs + node * (long long)(H * C) + (long long)h * C;
  const float* ar = att + (long long)h * C;
  float s = 0.f;
  for (int c = 0; c < C; c++) s += xr[c] * ar[c];
  out[i] = s;
}

// =================== edge kernels (E real edges + N self loops) ===================
__device__ __forceinline__ void edge_sd(const int* ei, long long e, int E, int* s, int* d) {
  if (e < E) { *s = ei[e]; *d = ei[(long long)E + e]; }
  else       { *s = *d = (int)(e - E); }
}
__device__ __forceinline__ float edge_score(const float* as, const float* ad,
                                            int s, int d, int H, int h) {
  float sc = as[(long long)s * H + h] + ad[(long long)d * H + h];
  return sc > 0.f ? sc : NEG_SLOPE_C * sc;
}

__global__ void k_edge_max(const int* __restrict__ ei, int E, int Nn, int H,
                           const float* __restrict__ as, const float* __restrict__ ad,
                           int* __restrict__ mb) {
  long long i = (long long)blockIdx.x * blockDim.x + threadIdx.x;
  long long tot = (long long)(E + Nn) * H;
  if (i >= tot) return;
  int h = (int)(i % H); long long e = i / H;
  int s, d; edge_sd(ei, e, E, &s, &d);
  float sc = edge_score(as, ad, s, d, H, h);
  atomicMax(mb + (long long)d * H + h, f2ord(sc));
}

__global__ void k_edge_sum(const int* __restrict__ ei, int E, int Nn, int H,
                           const float* __restrict__ as, const float* __restrict__ ad,
                           const int* __restrict__ mb, float* __restrict__ ssum) {
  long long i = (long long)blockIdx.x * blockDim.x + threadIdx.x;
  long long tot = (long long)(E + Nn) * H;
  if (i >= tot) return;
  int h = (int)(i % H); long long e = i / H;
  int s, d; edge_sd(ei, e, E, &s, &d);
  float sc = edge_score(as, ad, s, d, H, h);
  float m = ord2f(mb[(long long)d * H + h]);
  atomicAdd(ssum + (long long)d * H + h, expf(sc - m));
}

__global__ void k_edge_agg(const int* __restrict__ ei, int E, int Nn, int H, int C,
                           const float* __restrict__ as, const float* __restrict__ ad,
                           const int* __restrict__ mb, const float* __restrict__ ssum,
                           const float* __restrict__ xs, float* __restrict__ out) {
  long long i = (long long)blockIdx.x * blockDim.x + threadIdx.x;
  long long tot = (long long)(E + Nn) * H;
  if (i >= tot) return;
  int h = (int)(i % H); long long e = i / H;
  int s, d; edge_sd(ei, e, E, &s, &d);
  const float* xr = xs + (long long)s * H * C + (long long)h * C;
  __builtin_prefetch(xr, 0, 0);                       // global_prefetch_b8
  float sc = edge_score(as, ad, s, d, H, h);
  float m  = ord2f(mb[(long long)d * H + h]);
  float a  = expf(sc - m) / (ssum[(long long)d * H + h] + 1e-16f);
  float* orow = out + (long long)d * H * C + (long long)h * C;
  for (int c = 0; c < C; c++) atomicAdd(orow + c, xr[c] * a);
}

// =================== bias + batchnorm + ELU (in place) ===================
__global__ void k_bn_elu(float* __restrict__ hb, const float* __restrict__ bias,
                         const float* __restrict__ g, const float* __restrict__ b,
                         const float* __restrict__ m, const float* __restrict__ v,
                         int n, int D) {
  long long i = (long long)blockIdx.x * blockDim.x + threadIdx.x;
  if (i >= (long long)n * D) return;
  int d = (int)(i % D);
  float x = hb[i] + bias[d];
  x = (x - m[d]) * rsqrtf(v[d] + BN_EPS_C) * g[d] + b[d];
  hb[i] = x > 0.f ? x : expm1f(x);
}

// =================== graph pooling ===================
__global__ void k_pool(const float* __restrict__ hb, const int* __restrict__ batch,
                       int n, int D, float* __restrict__ gsum, int* __restrict__ gmaxb,
                       float* __restrict__ gcnt) {
  int i = blockIdx.x * blockDim.x + threadIdx.x;
  if (i >= n) return;
  int bg = batch[i];
  atomicAdd(gcnt + bg, 1.0f);
  const float* hr = hb + (long long)i * D;
  float* gs = gsum + (long long)bg * D;
  int*   gm = gmaxb + (long long)bg * D;
  for (int d = 0; d < D; d++) {
    float hv = hr[d];
    atomicAdd(gs + d, hv);
    atomicMax(gm + d, f2ord(hv));
  }
}

// =================== classifier MLP ===================
__global__ void k_mlp(const float* __restrict__ gsum, const int* __restrict__ gmaxb,
                      const float* __restrict__ gcnt, const float* __restrict__ u,
                      const float* __restrict__ W1, const float* __restrict__ b1,
                      const float* __restrict__ W2, const float* __restrict__ b2,
                      const float* __restrict__ W3, const float* __restrict__ b3,
                      float* __restrict__ out, int G, int D) {
  int g = blockIdx.x * blockDim.x + threadIdx.x;
  if (g >= G) return;
  const int COMB = 2 * HID_C + GLOB_F_C;  // 74
  float z[COMB];
  float cnt = gcnt[g];
  float ic = 1.f / fmaxf(cnt, 1.f);
  for (int d = 0; d < D; d++) {
    z[d] = gsum[(long long)g * D + d] * ic;
    float mx = ord2f(gmaxb[(long long)g * D + d]);
    z[D + d] = cnt > 0.f ? mx : 0.f;
  }
  for (int j = 0; j < GLOB_F_C; j++) z[2 * D + j] = u[(long long)g * GLOB_F_C + j];
  float z1[HID_C];
  for (int o = 0; o < HID_C; o++) {
    float a = b1[o];
    for (int k = 0; k < COMB; k++) a += z[k] * W1[k * HID_C + o];
    z1[o] = fmaxf(a, 0.f);
  }
  float z2[HID_C / 2];
  for (int o = 0; o < HID_C / 2; o++) {
    float a = b2[o];
    for (int k = 0; k < HID_C; k++) a += z1[k] * W2[k * (HID_C / 2) + o];
    z2[o] = fmaxf(a, 0.f);
  }
  float a = b3[0];
  for (int k = 0; k < HID_C / 2; k++) a += z2[k] * W3[k];
  out[g] = 1.f / (1.f + expf(-a));
}

// =================== host ===================
extern "C" void kernel_launch(void* const* d_in, const int* in_sizes, int n_in,
                              void* d_out, int out_size, void* d_ws, size_t ws_size,
                              hipStream_t stream) {
  (void)out_size; (void)ws_size;
  const float *x, *u, *W1, *b1, *W2, *b2, *W3, *b3;
  const int *edge_index, *batch;
  const float *Wl[3], *asrc[3], *adst[3], *bias[3], *bng[3], *bnb[3], *bnm[3], *bnv[3];

  // JAX pytree flatten sorts dict keys -> batch is leaf 0 (size 100000).
  // Fallback: insertion-order flatten -> x is leaf 0 (size 400000).
  bool sortedOrder = (n_in > 0 && in_sizes[0] == N_NODES_C);
  if (sortedOrder) {
    batch      = (const int*)d_in[0];
    edge_index = (const int*)d_in[1];
    W1 = (const float*)d_in[2]; W2 = (const float*)d_in[3]; W3 = (const float*)d_in[4];
    b1 = (const float*)d_in[5]; b2 = (const float*)d_in[6]; b3 = (const float*)d_in[7];
    for (int i = 0; i < 3; i++) {
      int base = 8 + 8 * i;  // sorted: W, att_dst, att_src, bias, bn_b, bn_g, bn_m, bn_v
      Wl[i]   = (const float*)d_in[base + 0];
      adst[i] = (const float*)d_in[base + 1];
      asrc[i] = (const float*)d_in[base + 2];
      bias[i] = (const float*)d_in[base + 3];
      bnb[i]  = (const float*)d_in[base + 4];
      bng[i]  = (const float*)d_in[base + 5];
      bnm[i]  = (const float*)d_in[base + 6];
      bnv[i]  = (const float*)d_in[base + 7];
    }
    u = (const float*)d_in[32];
    x = (const float*)d_in[33];
  } else {
    x = (const float*)d_in[0];
    u = (const float*)d_in[1];
    for (int i = 0; i < 3; i++) {
      int base = 2 + 8 * i;  // insertion: W, att_src, att_dst, bias, bn_g, bn_b, bn_m, bn_v
      Wl[i]   = (const float*)d_in[base + 0];
      asrc[i] = (const float*)d_in[base + 1];
      adst[i] = (const float*)d_in[base + 2];
      bias[i] = (const float*)d_in[base + 3];
      bng[i]  = (const float*)d_in[base + 4];
      bnb[i]  = (const float*)d_in[base + 5];
      bnm[i]  = (const float*)d_in[base + 6];
      bnv[i]  = (const float*)d_in[base + 7];
    }
    W1 = (const float*)d_in[26]; b1 = (const float*)d_in[27];
    W2 = (const float*)d_in[28]; b2 = (const float*)d_in[29];
    W3 = (const float*)d_in[30]; b3 = (const float*)d_in[31];
    edge_index = (const int*)d_in[32];
    batch      = (const int*)d_in[33];
  }

  // ---- workspace carve (256B aligned) ----
  char* wsp = (char*)d_ws;
  size_t off = 0;
  auto carve = [&](size_t bytes) -> void* {
    void* p = wsp + off;
    off += (bytes + 255) & ~(size_t)255;
    return p;
  };
  float*    xsBuf = (float*)carve((size_t)N_NODES_C * 128 * sizeof(float));
  float*    hBuf  = (float*)carve((size_t)N_NODES_C * 128 * sizeof(float));
  _Float16* A16   = (_Float16*)carve((size_t)N_NODES_C * 128 * sizeof(_Float16));
  _Float16* Wt16  = (_Float16*)carve((size_t)128 * 128 * sizeof(_Float16));
  float*    a_s   = (float*)carve((size_t)N_NODES_C * HEADS_C * sizeof(float));
  float*    a_d   = (float*)carve((size_t)N_NODES_C * HEADS_C * sizeof(float));
  int*      mb    = (int*)carve((size_t)N_NODES_C * HEADS_C * sizeof(int));
  float*    ssum  = (float*)carve((size_t)N_NODES_C * HEADS_C * sizeof(float));
  float*    gsum  = (float*)carve((size_t)N_GRAPHS_C * HID_C * sizeof(float));
  int*      gmaxb = (int*)carve((size_t)N_GRAPHS_C * HID_C * sizeof(int));
  float*    gcnt  = (float*)carve((size_t)N_GRAPHS_C * sizeof(float));

  const int T = 256;
  auto blocks = [](long long n, int t) { return (int)((n + t - 1) / t); };

  struct Cfg { int inK, padK, H, C, outD; };
  Cfg cfg[3] = {
      {NODE_F_C,          32,  HEADS_C, HID_C, HEADS_C * HID_C},  // 4 -> 128
      {HEADS_C * HID_C,  128,  HEADS_C, HID_C, HEADS_C * HID_C},  // 128 -> 128
      {HEADS_C * HID_C,  128,  1,       HID_C, HID_C},            // 128 -> 32
  };

  const int tilesM = N_NODES_C / 16;  // N_NODES_C % 16 == 0
  const int gx = (tilesM + 7) / 8;

  const float* hin = x;
  for (int i = 0; i < 3; i++) {
    const int inK = cfg[i].inK, padK = cfg[i].padK, H = cfg[i].H, C = cfg[i].C, D = cfg[i].outD;
    const long long na = (long long)N_NODES_C * padK;
    const long long nd = (long long)N_NODES_C * D;
    const int nh = N_NODES_C * H;

    // feature transform: xs = h @ W  (f16 WMMA, f32 accumulate, B staged in LDS)
    k_cvt_a<<<blocks(na, T), T, 0, stream>>>(hin, A16, N_NODES_C, inK, padK);
    k_cvt_w<<<blocks((long long)D * padK, T), T, 0, stream>>>(Wl[i], Wt16, inK, D, padK);
    if (i == 0) {
      k_gemm_wmma<4, 32, 128><<<dim3(gx, 2), T, 0, stream>>>(A16, Wt16, xsBuf, N_NODES_C);
    } else if (i == 1) {
      k_gemm_wmma<4, 128, 128><<<dim3(gx, 2), T, 0, stream>>>(A16, Wt16, xsBuf, N_NODES_C);
    } else {
      k_gemm_wmma<2, 128, 32><<<dim3(gx, 1), T, 0, stream>>>(A16, Wt16, xsBuf, N_NODES_C);
    }

    // per-node attention score halves
    k_att<<<blocks(nh, T), T, 0, stream>>>(xsBuf, asrc[i], a_s, N_NODES_C, H, C);
    k_att<<<blocks(nh, T), T, 0, stream>>>(xsBuf, adst[i], a_d, N_NODES_C, H, C);

    // segment softmax + scatter aggregation over E+N edges (self loops appended)
    k_fill_i32<<<blocks(nh, T), T, 0, stream>>>(mb, nh, ORD_NEG_INF);
    k_fill_f32<<<blocks(nh, T), T, 0, stream>>>(ssum, nh, 0.f);
    k_fill_f32<<<blocks(nd, T), T, 0, stream>>>(hBuf, nd, 0.f);
    const long long eth = (long long)(N_EDGES_C + N_NODES_C) * H;
    k_edge_max<<<blocks(eth, T), T, 0, stream>>>(edge_index, N_EDGES_C, N_NODES_C, H, a_s, a_d, mb);
    k_edge_sum<<<blocks(eth, T), T, 0, stream>>>(edge_index, N_EDGES_C, N_NODES_C, H, a_s, a_d, mb, ssum);
    k_edge_agg<<<blocks(eth, T), T, 0, stream>>>(edge_index, N_EDGES_C, N_NODES_C, H, C,
                                                 a_s, a_d, mb, ssum, xsBuf, hBuf);

    // bias + batchnorm + ELU (in place on hBuf)
    k_bn_elu<<<blocks(nd, T), T, 0, stream>>>(hBuf, bias[i], bng[i], bnb[i], bnm[i], bnv[i],
                                              N_NODES_C, D);
    hin = hBuf;
  }

  // ---- graph pooling (mean + max) ----
  k_fill_f32<<<blocks((long long)N_GRAPHS_C * HID_C, T), T, 0, stream>>>(gsum, N_GRAPHS_C * HID_C, 0.f);
  k_fill_i32<<<blocks((long long)N_GRAPHS_C * HID_C, T), T, 0, stream>>>(gmaxb, N_GRAPHS_C * HID_C, ORD_NEG_INF);
  k_fill_f32<<<blocks(N_GRAPHS_C, T), T, 0, stream>>>(gcnt, N_GRAPHS_C, 0.f);
  k_pool<<<blocks(N_NODES_C, T), T, 0, stream>>>(hBuf, batch, N_NODES_C, HID_C, gsum, gmaxb, gcnt);

  // ---- classifier ----
  k_mlp<<<blocks(N_GRAPHS_C, T), T, 0, stream>>>(gsum, gmaxb, gcnt, u, W1, b1, W2, b2, W3, b3,
                                                 (float*)d_out, N_GRAPHS_C, HID_C);
}
